// capsnet_10934986736334
// MI455X (gfx1250) — compile-verified, matrix operand order checked by reading, not run on hardware
//
#include <hip/hip_runtime.h>

typedef __attribute__((ext_vector_type(16))) _Float16 v16h;
typedef __attribute__((ext_vector_type(8)))  _Float16 v8h;
typedef __attribute__((ext_vector_type(4)))  _Float16 v4h;
typedef __attribute__((ext_vector_type(2)))  _Float16 v2h;
typedef __attribute__((ext_vector_type(8)))  float    v8f;

#define IN_CAPS 8192
#define NBATCH  64
#define JM      128   // out_caps(8) * out_dim(16)

__device__ inline v16h zero16h() {
  v16h r;
  #pragma unroll
  for (int e = 0; e < 16; ++e) r[e] = (_Float16)0.0f;
  return r;
}

// ---------------------------------------------------------------------------
// Prep kernels: one-time fp32 -> fp16 conversion into WMMA-operand layouts.
//   Wp[((i*8+j)*16+m)*8 + k] = W[i][j][k][m]   (B operand: lane m loads b128)
//   xh[b][i][k]              = x   (A operand: lane loads one b128)
// ---------------------------------------------------------------------------
__global__ __launch_bounds__(256)
void prep_w_kernel(const float* __restrict__ W, _Float16* __restrict__ Wp) {
  const int t  = blockIdx.x * 256 + threadIdx.x;   // t = (i*8+j)*16 + m, < 1048576
  const int m  = t & 15;
  const int ij = t >> 4;
  const float* wp = W + (size_t)ij * 128 + m;      // k stride = 16 floats
  v8h o;
  #pragma unroll
  for (int k = 0; k < 8; ++k) o[k] = (_Float16)wp[k * 16];
  *(v8h*)(Wp + (size_t)t * 8) = o;                 // 16B contiguous store
}

__global__ __launch_bounds__(256)
void prep_x_kernel(const float* __restrict__ x, _Float16* __restrict__ xh) {
  const int t = blockIdx.x * 256 + threadIdx.x;    // 4 elements each, < 1048576
  const float4 v = ((const float4*)x)[t];
  v4h o;
  o[0] = (_Float16)v.x; o[1] = (_Float16)v.y;
  o[2] = (_Float16)v.z; o[3] = (_Float16)v.w;
  *(v4h*)(xh + (size_t)t * 4) = o;
}

// ---------------------------------------------------------------------------
// Kernel: u_hat[b,i,j,m] = sum_n x[b,i,n] * W[i,j,n,m]  via WMMA.
// Per i: (64 x 8) x (8 x 128) GEMM. M = batch tile of 16, N = one out-cap j
// (m across lanes), K=8 zero-padded in the K=32 slot (fp32 accumulate).
// Hot loop: 1 b128 A-load per i, 1 b128 B-load + 1 wmma + 8 packed b16
// stores per (i,j). Grid: 512 blocks x 8 waves; wave -> (b-tile, i-chunk of 8).
// ---------------------------------------------------------------------------
__global__ __launch_bounds__(256)
void uhat_wmma_kernel(const _Float16* __restrict__ xh,
                      const _Float16* __restrict__ Wp,
                      _Float16* __restrict__ uhat) {
  const int lane = threadIdx.x & 31;
  const int wv   = threadIdx.x >> 5;
  const int gw   = blockIdx.x * 8 + wv;     // 0..4095
  const int bt   = gw & 3;                  // batch tile (16 rows of 64)
  const int ic   = gw >> 2;                 // i-chunk index (8 i's)
  const int row  = lane & 15;
  const bool lo  = lane < 16;               // lanes 16..31 hold K=8..15 -> zero

  for (int ii = 0; ii < 8; ++ii) {
    const int i = ic * 8 + ii;

    v16h a = zero16h();
    if (lo) {
      const v8h t = *(const v8h*)(xh + ((size_t)(bt * 16 + row) * IN_CAPS + i) * 8);
      #pragma unroll
      for (int e = 0; e < 8; ++e) a[e] = t[e];
    }

    #pragma unroll
    for (int j = 0; j < 8; ++j) {
      v16h bm = zero16h();
      if (lo) {
        const v8h t = *(const v8h*)(Wp + (((size_t)i * 8 + j) * 16 + row) * 8);
        #pragma unroll
        for (int e = 0; e < 8; ++e) bm[e] = t[e];
      }
      v8f acc = {};
      acc = __builtin_amdgcn_wmma_f32_16x16x32_f16(
          false, a, false, bm, (short)0, acc, false, false);

      // D layout: VGPR r -> (M=r, N=lane) lanes 0-15; (M=8+r, N=lane-16) lanes 16-31
      const int brow = bt * 16 + (lo ? 0 : 8);
      _Float16* up = uhat + (((size_t)brow * IN_CAPS + i) * 8 + j) * 16 + row;
      #pragma unroll
      for (int r = 0; r < 8; ++r)
        up[(size_t)r * ((size_t)IN_CAPS * JM)] = (_Float16)acc[r];
    }
  }
}

// ---------------------------------------------------------------------------
// Routing pass kernels. Grid = 512 WGs: (b 0..63) x (i-slice 0..7), 256 thr.
// Each pass streams u_hat[b, slice*1024 .. +1023] once. Groups of 4 lanes own
// one i (each lane: 2 out-caps -> 64B contiguous coalesced loads); softmax
// over the 8 j's via __shfl_xor within the 4-lane group. Per-WG partial s is
// reduced in fixed order through LDS (no atomics -> bitwise deterministic)
// and written to out_part[b][slice][jm]. MODE: 0 = uniform c (round 0, no
// logit I/O); 1 = first routing round (logits write-only); 2 = read+update.
// ---------------------------------------------------------------------------
__device__ inline float squash_factor(float sq) {
  return sq / ((1.0f + sq) * sqrtf(sq));
}

template <int MODE>
__global__ __launch_bounds__(256)
void route_pass_kernel(const _Float16* __restrict__ uhat,
                       _Float16* __restrict__ logits,
                       const float* __restrict__ prev_part,
                       float* __restrict__ out_part) {
  __shared__ float v_sh[JM];
  __shared__ float red[64 * JM];             // 32 KB fixed-order reduction

  const int blk   = blockIdx.x;              // 0..511
  const int b     = blk >> 3;
  const int slice = blk & 7;
  const int tid   = threadIdx.x;
  const int sub   = tid & 3;                 // j-pair: j0=2*sub, j1=2*sub+1
  const int grp   = tid >> 2;                // 0..63

  if constexpr (MODE != 0) {
    if (tid < JM) {                          // tid = j*16+m, 4 full waves
      float s = 0.0f;
      #pragma unroll
      for (int sl = 0; sl < 8; ++sl)
        s += prev_part[((size_t)b * 8 + sl) * JM + tid];
      float sq = s * s;
      sq += __shfl_xor(sq, 1); sq += __shfl_xor(sq, 2);
      sq += __shfl_xor(sq, 4); sq += __shfl_xor(sq, 8);
      v_sh[tid] = s * squash_factor(sq);
    }
    __syncthreads();
  }

  const _Float16* ub = uhat + (size_t)b * IN_CAPS * JM;
  float sacc[32];
  #pragma unroll
  for (int e = 0; e < 32; ++e) sacc[e] = 0.0f;

  for (int k = 0; k < 16; ++k) {
    const int i = slice * 1024 + grp + 64 * k;
    const v16h* uv = (const v16h*)(ub + (size_t)i * JM + sub * 32);
    const v16h u0 = uv[0], u1 = uv[1];
    float u[32];
    #pragma unroll
    for (int m = 0; m < 16; ++m) {
      u[m]      = (float)u0[m];
      u[16 + m] = (float)u1[m];
    }

    float c0, c1;
    if constexpr (MODE == 0) {
      c0 = 0.125f; c1 = 0.125f;
    } else {
      float a0 = 0.0f, a1 = 0.0f;            // agreement <v[j,:], u[i,j,:]>
      #pragma unroll
      for (int m = 0; m < 16; ++m) {
        a0 += v_sh[sub * 32 + m]      * u[m];
        a1 += v_sh[sub * 32 + 16 + m] * u[16 + m];
      }
      v2h* lp = (v2h*)(logits + ((size_t)b * IN_CAPS + i) * 8 + sub * 2);
      float l0 = a0, l1 = a1;
      if constexpr (MODE == 2) {
        const v2h old = *lp;
        l0 += (float)old[0];
        l1 += (float)old[1];
      }
      v2h nw; nw[0] = (_Float16)l0; nw[1] = (_Float16)l1;
      *lp = nw;
      // softmax over 8 j's across the 4-lane group
      float mx = fmaxf(l0, l1);
      mx = fmaxf(mx, __shfl_xor(mx, 1));
      mx = fmaxf(mx, __shfl_xor(mx, 2));
      const float e0 = __expf(l0 - mx), e1 = __expf(l1 - mx);
      float es = e0 + e1;
      es += __shfl_xor(es, 1);
      es += __shfl_xor(es, 2);
      const float inv = 1.0f / es;
      c0 = e0 * inv; c1 = e1 * inv;
    }

    #pragma unroll
    for (int m = 0; m < 16; ++m) {
      sacc[m]      += c0 * u[m];
      sacc[16 + m] += c1 * u[16 + m];
    }
  }

  // deterministic fixed-order reduction across the 64 groups
  #pragma unroll
  for (int e = 0; e < 32; ++e) red[grp * JM + sub * 32 + e] = sacc[e];
  __syncthreads();
  if (tid < JM) {
    float s = 0.0f;
    for (int g = 0; g < 64; ++g) s += red[g * JM + tid];
    out_part[((size_t)b * 8 + slice) * JM + tid] = s;
  }
}

// Final reduce + squash -> output [b, j, m]
__global__ __launch_bounds__(128)
void route_out_kernel(const float* __restrict__ part, float* __restrict__ out) {
  const int b = blockIdx.x, tid = threadIdx.x;   // tid = j*16+m
  float s = 0.0f;
  #pragma unroll
  for (int sl = 0; sl < 8; ++sl)
    s += part[((size_t)b * 8 + sl) * JM + tid];
  float sq = s * s;
  sq += __shfl_xor(sq, 1); sq += __shfl_xor(sq, 2);
  sq += __shfl_xor(sq, 4); sq += __shfl_xor(sq, 8);
  out[(size_t)b * JM + tid] = s * squash_factor(sq);
}

// ---------------------------------------------------------------------------
extern "C" void kernel_launch(void* const* d_in, const int* in_sizes, int n_in,
                              void* d_out, int out_size, void* d_ws, size_t ws_size,
                              hipStream_t stream) {
  const float* x = (const float*)d_in[0];   // [64, 8192, 8]
  const float* W = (const float*)d_in[1];   // [8192, 8, 8, 16]
  float* out = (float*)d_out;               // [64, 8, 16]

  char* ws = (char*)d_ws;
  _Float16* uhat   = (_Float16*)(ws);                       // 134,217,728 B
  _Float16* Wp     = (_Float16*)(ws + 134217728ull);        //  16,777,216 B
  _Float16* xh     = (_Float16*)(ws + 150994944ull);        //   8,388,608 B
  _Float16* logits = (_Float16*)(ws + 159383552ull);        //   8,388,608 B
  float*    part_a = (float*)(ws + 167772160ull);           //     262,144 B
  float*    part_b = (float*)(ws + 168034304ull);           //     262,144 B

  prep_w_kernel<<<4096, 256, 0, stream>>>(W, Wp);
  prep_x_kernel<<<4096, 256, 0, stream>>>(x, xh);
  uhat_wmma_kernel<<<512, 256, 0, stream>>>(xh, Wp, uhat);

  route_pass_kernel<0><<<512, 256, 0, stream>>>(uhat, logits, nullptr, part_a);
  route_pass_kernel<1><<<512, 256, 0, stream>>>(uhat, logits, part_a, part_b);
  route_pass_kernel<2><<<512, 256, 0, stream>>>(uhat, logits, part_b, part_a);
  route_out_kernel<<<NBATCH, 128, 0, stream>>>(part_a, out);
}
